// PhiMoESparseMoeBlock_1967095021957
// MI455X (gfx1250) — matrix operand order repaired and checked by reading, compiled, not run on hardware
//
#include <hip/hip_runtime.h>
#include <math.h>

// Problem constants (match reference)
#define Bd 4
#define Sd 2048
#define Hd 1024
#define Fd 4096
#define Ed 8
#define Nd (Bd * Sd)          // 8192 tokens
#define JEPS 0.01f

#define TM 16                  // tokens per tile
#define TF 64                  // F-chunk width

typedef __attribute__((ext_vector_type(16))) __bf16 v16bf;
typedef __attribute__((ext_vector_type(8)))  __bf16 v8bf;
typedef __attribute__((ext_vector_type(8)))  float  v8f;

// ---------------------------------------------------------------------------
// A-fragment (16x32 bf16) in contiguous K order (this code's choice, same as B): lane m = lane&15;
// lanes 0-15 hold K = k0+0..15, lanes 16-31 hold K = k0+16..31.
__device__ inline v16bf load_a_frag(const __bf16* p) {
    v8bf lo = *(const v8bf*)(p);
    v8bf hi = *(const v8bf*)(p + 8);
    v16bf a;
#pragma unroll
    for (int i = 0; i < 8; ++i) { a[i] = lo[i]; a[i + 8] = hi[i]; }
    return a;
}

// B-fragment (32x16 bf16): lane n = lane&15; lanes 0-15 hold K=0..15,
// lanes 16-31 hold K=16..31 of column n.
__device__ inline v16bf load_b_frag(const float* p) {
    float4 q0 = ((const float4*)p)[0];
    float4 q1 = ((const float4*)p)[1];
    float4 q2 = ((const float4*)p)[2];
    float4 q3 = ((const float4*)p)[3];
    v16bf b;
    b[0]  = (__bf16)q0.x; b[1]  = (__bf16)q0.y; b[2]  = (__bf16)q0.z; b[3]  = (__bf16)q0.w;
    b[4]  = (__bf16)q1.x; b[5]  = (__bf16)q1.y; b[6]  = (__bf16)q1.z; b[7]  = (__bf16)q1.w;
    b[8]  = (__bf16)q2.x; b[9]  = (__bf16)q2.y; b[10] = (__bf16)q2.z; b[11] = (__bf16)q2.w;
    b[12] = (__bf16)q3.x; b[13] = (__bf16)q3.y; b[14] = (__bf16)q3.z; b[15] = (__bf16)q3.w;
    return b;
}
__device__ inline v16bf load_b_frag(const __bf16* p) {
    return *(const v16bf*)p;     // 32B/lane, no VALU
}

__device__ inline float fast_rcp(float x) { return __builtin_amdgcn_rcpf(x); }

// ---------------------------------------------------------------------------
// One-shot fp32 -> bf16 weight conversion (8 elements/thread).
__global__ __launch_bounds__(256) void convert_w_kernel(
    const float* __restrict__ src, __bf16* __restrict__ dst)
{
    size_t i = ((size_t)blockIdx.x * 256 + threadIdx.x) * 8;
    float4 a = *(const float4*)(src + i);
    float4 b = *(const float4*)(src + i + 4);
    v8bf o;
    o[0] = (__bf16)a.x; o[1] = (__bf16)a.y; o[2] = (__bf16)a.z; o[3] = (__bf16)a.w;
    o[4] = (__bf16)b.x; o[5] = (__bf16)b.y; o[6] = (__bf16)b.z; o[7] = (__bf16)b.w;
    *(v8bf*)(dst + i) = o;
}

// ---------------------------------------------------------------------------
// Router: one wave per token; sparse-mixer top-2; scatter to per-expert lists.
__global__ __launch_bounds__(256) void moe_router_kernel(
    const float* __restrict__ x, const float* __restrict__ gate_w,
    int* __restrict__ counts, int* __restrict__ tokIdx, float* __restrict__ tokW)
{
    __shared__ float gw[Ed * Hd];
    for (int i = threadIdx.x; i < Ed * Hd; i += 256) gw[i] = gate_w[i];
    __syncthreads();

    const int lane = threadIdx.x & 31;
    const int wave = threadIdx.x >> 5;
    const int t = blockIdx.x * 8 + wave;
    const float* xr = x + (size_t)t * Hd;

    float s[Ed];
#pragma unroll
    for (int e = 0; e < Ed; ++e) s[e] = 0.f;
    for (int k = lane; k < Hd; k += 32) {
        float xv = xr[k];
#pragma unroll
        for (int e = 0; e < Ed; ++e) s[e] = fmaf(xv, gw[e * Hd + k], s[e]);
    }
#pragma unroll
    for (int e = 0; e < Ed; ++e) {
#pragma unroll
        for (int off = 16; off > 0; off >>= 1) s[e] += __shfl_xor(s[e], off, 32);
    }

    if (lane == 0) {
        int i0 = 0; float m1 = s[0];
        for (int e = 1; e < Ed; ++e) if (s[e] > m1) { m1 = s[e]; i0 = e; }
        int i1 = -1; float m2 = -3.402823466e38f;
        for (int e = 0; e < Ed; ++e) if (e != i0 && s[e] > m2) { m2 = s[e]; i1 = e; }

        float den1 = 0.f, num1 = 0.f;
        for (int e = 0; e < Ed; ++e) {
            float factor = fmaxf(fabsf(s[e]), m1);
            bool mask = ((m1 - s[e]) * fast_rcp(factor)) > (2.f * JEPS);
            float ev = mask ? 0.f : __expf(s[e] - m1);
            den1 += ev;
            if (e == i0) num1 = ev;
        }
        float mult1 = num1 * fast_rcp(den1);

        float den2 = 0.f, num2 = 0.f;
        for (int e = 0; e < Ed; ++e) {
            float factor = fmaxf(fabsf(s[e]), m2);
            bool mask = (((m2 - s[e]) * fast_rcp(factor)) > (2.f * JEPS)) || (e == i0);
            float ev = mask ? 0.f : __expf(s[e] - m2);
            den2 += ev;
            if (e == i1) num2 = ev;
        }
        float mult2 = num2 * fast_rcp(den2);

        int slot0 = atomicAdd(&counts[i0], 1);
        tokIdx[i0 * Nd + slot0] = t; tokW[i0 * Nd + slot0] = mult1;
        int slot1 = atomicAdd(&counts[i1], 1);
        tokIdx[i1 * Nd + slot1] = t; tokW[i1 * Nd + slot1] = mult2;
    }
}

// ---------------------------------------------------------------------------
// Grouped expert GEMM, fused SwiGLU. Block = 256 threads = 8 waves, one
// 16-token tile of one expert. Output tile (16x1024 f32) lives in registers.
// WT = __bf16 (pre-converted weights, preferred) or float (inline convert).
template <typename WT>
__global__ __launch_bounds__(256) void moe_expert_kernel(
    const float* __restrict__ x,
    const WT* __restrict__ w1, const WT* __restrict__ w2, const WT* __restrict__ w3,
    const int* __restrict__ counts, const int* __restrict__ tokIdx, const float* __restrict__ tokW,
    float* __restrict__ out)
{
    const int e = blockIdx.y;
    const int cnt = counts[e];
    const int t0 = blockIdx.x * TM;
    if (t0 >= cnt) return;

    __shared__ float  xstage[4 * Hd];   // async-load staging, 16 KB
    __shared__ __bf16 xs[TM * Hd];      // gathered X tile (bf16), 32 KB
    __shared__ float  acc1s[TM * TF];   // x @ W1^T chunk, 4 KB
    __shared__ float  acc3s[TM * TF];   // x @ W3^T chunk, 4 KB
    __shared__ __bf16 hs[TM * TF];      // silu(a1)*a3 (bf16), 2 KB
    __shared__ int    tok_s[TM];
    __shared__ float  wt_s[TM];

    const int tid  = threadIdx.x;
    const int lane = tid & 31;
    const int wave = tid >> 5;

    if (tid < TM) {
        int i = t0 + tid;
        tok_s[tid] = (i < cnt) ? tokIdx[e * Nd + i] : tokIdx[e * Nd];
        wt_s[tid]  = (i < cnt) ? tokW[e * Nd + i] : 0.f;
    }
    __syncthreads();

    // Gather 16 token rows of X: async global->LDS (b128 chunks) into fp32
    // staging, then convert to bf16 tile. 4 rows per pass.
    for (int rg = 0; rg < TM; rg += 4) {
#pragma unroll
        for (int i = tid; i < 4 * 256; i += 256) {        // 256 b128 chunks/row
            const int r = i >> 8;
            const int c = i & 255;
            unsigned long long ga =
                (unsigned long long)(uintptr_t)(x + (size_t)tok_s[rg + r] * Hd + c * 4);
            unsigned int la = (unsigned int)(uintptr_t)&xstage[r * Hd + c * 4];
            asm volatile("global_load_async_to_lds_b128 %0, %1, off"
                         :: "v"(la), "v"(ga) : "memory");
        }
        asm volatile("s_wait_asynccnt 0x0" ::: "memory");
        __syncthreads();
        for (int i = tid; i < 4 * Hd; i += 256)
            xs[rg * Hd + i] = (__bf16)xstage[i];
        __syncthreads();
    }

    const int mA    = lane & 15;               // A-fragment row
    const int koffA = (lane < 16) ? 0 : 16;    // A-fragment K sub-offset
    const int nB    = lane & 15;               // B-fragment column
    const int kbB   = (lane < 16) ? 0 : 16;    // B-fragment K base
    const int mbase = (lane < 16) ? 0 : 8;     // C-fragment row base
    const size_t eFH = (size_t)e * Fd * Hd;

    v8f outacc[8];
#pragma unroll
    for (int i = 0; i < 8; ++i) outacc[i] = {};

    for (int f0 = 0; f0 < Fd; f0 += TF) {
        // ---- stage 1: waves 0-3 -> acc1 (W1), waves 4-7 -> acc3 (W3)
        const WT*  wsel   = (wave < 4) ? w1 : w3;
        float*     accbuf = (wave < 4) ? acc1s : acc3s;
        const int  ntile  = wave & 3;
        const WT*  wrow   = wsel + eFH + (size_t)(f0 + ntile * 16 + nB) * Hd + kbB;

        v8f acc = {};
#pragma unroll 4
        for (int k0 = 0; k0 < Hd; k0 += 32) {
            v16bf a = load_a_frag(&xs[mA * Hd + k0 + koffA]);
            v16bf b = load_b_frag(wrow + k0);
            acc = __builtin_amdgcn_wmma_f32_16x16x32_bf16(
                false, a, false, b, (short)0, acc, false, false);
        }
#pragma unroll
        for (int v = 0; v < 8; ++v)
            accbuf[(mbase + v) * TF + ntile * 16 + (lane & 15)] = acc[v];
        __syncthreads();

        // ---- SwiGLU: h = a * sigmoid(a) * b3, via v_rcp_f32 (no IEEE divide)
        for (int i = tid; i < TM * TF; i += 256) {
            float a  = acc1s[i];
            float b3 = acc3s[i];
            float sig = fast_rcp(1.f + __expf(-a));
            hs[i] = (__bf16)(a * sig * b3);
        }
        __syncthreads();

        // ---- stage 2: out[16 x 1024] += h[16 x 64] @ W2[:, f0:f0+64]^T
#pragma unroll
        for (int nt = 0; nt < 8; ++nt) {
            const int n0 = wave * 128 + nt * 16;
            const WT* w2row = w2 + (size_t)e * Hd * Fd + (size_t)(n0 + nB) * Fd + f0 + kbB;
#pragma unroll
            for (int k0 = 0; k0 < TF; k0 += 32) {
                v16bf a = load_a_frag(&hs[mA * TF + k0 + koffA]);
                v16bf b = load_b_frag(w2row + k0);
                outacc[nt] = __builtin_amdgcn_wmma_f32_16x16x32_bf16(
                    false, a, false, b, (short)0, outacc[nt], false, false);
            }
        }
        __syncthreads();
    }

    // ---- scaled scatter-add (token appears in two experts -> atomics)
#pragma unroll
    for (int nt = 0; nt < 8; ++nt) {
        const int col = wave * 128 + nt * 16 + (lane & 15);
#pragma unroll
        for (int v = 0; v < 8; ++v) {
            const int m = mbase + v;
            atomicAdd(&out[(size_t)tok_s[m] * Hd + col], wt_s[m] * outacc[nt][v]);
        }
    }
}

// ---------------------------------------------------------------------------
extern "C" void kernel_launch(void* const* d_in, const int* in_sizes, int n_in,
                              void* d_out, int out_size, void* d_ws, size_t ws_size,
                              hipStream_t stream) {
    const float* x      = (const float*)d_in[0];
    const float* gate_w = (const float*)d_in[1];
    const float* w1     = (const float*)d_in[2];
    const float* w2     = (const float*)d_in[3];
    const float* w3     = (const float*)d_in[4];
    float* out = (float*)d_out;

    char* ws = (char*)d_ws;
    int*   counts = (int*)ws;                              // 8 ints (64B slot)
    int*   tokIdx = (int*)(ws + 64);                       // E*N ints (256 KB)
    float* tokW   = (float*)(ws + 64 + (size_t)Ed * Nd * sizeof(int));

    const size_t routeBytes = 64 + (size_t)Ed * Nd * 4 * 2;        // 512 KB + 64
    const size_t wElems     = (size_t)Ed * Fd * Hd;                // 33.5 M
    const bool   useBf16W   = ws_size >= routeBytes + 3 * wElems * sizeof(__bf16) + 256;

    hipMemsetAsync(counts, 0, 64, stream);
    hipMemsetAsync(d_out, 0, (size_t)out_size * sizeof(float), stream);

    moe_router_kernel<<<Nd / 8, 256, 0, stream>>>(x, gate_w, counts, tokIdx, tokW);

    dim3 grid(Nd / TM, Ed);   // worst case: all tokens on one expert
    if (useBf16W) {
        __bf16* w1b = (__bf16*)(ws + routeBytes);
        __bf16* w3b = w1b + wElems;
        __bf16* w2b = w3b + wElems;
        const int cvtBlocks = (int)(wElems / (8 * 256));           // exact: 16384
        convert_w_kernel<<<cvtBlocks, 256, 0, stream>>>(w1, w1b);
        convert_w_kernel<<<cvtBlocks, 256, 0, stream>>>(w3, w3b);
        convert_w_kernel<<<cvtBlocks, 256, 0, stream>>>(w2, w2b);
        moe_expert_kernel<__bf16><<<grid, 256, 0, stream>>>(
            x, w1b, w2b, w3b, counts, tokIdx, tokW, out);
    } else {
        moe_expert_kernel<float><<<grid, 256, 0, stream>>>(
            x, w1, w2, w3, counts, tokIdx, tokW, out);
    }
}
